// Stochasticity_51230369906920
// MI455X (gfx1250) — matrix operand order repaired and checked by reading, not compiled, not run on hardware
//
#include <hip/hip_runtime.h>
#include <hip/hip_bf16.h>

// ---------------------------------------------------------------------------
// Shapes
// ---------------------------------------------------------------------------
#define Bq   256
#define Tq   64
#define Lq   16
#define Eq   64
#define Hq   256
#define Rq   512
#define LEq  64
#define NLq  17
#define Nq   (Bq*Tq)        // 16384 segment rows
#define G4H  (4*Hq)         // 1024
#define G3R  (3*Rq)         // 1536
#define DIN  (Rq+LEq)       // 576

typedef __attribute__((ext_vector_type(8)))  __bf16 v8bf;
typedef __attribute__((ext_vector_type(16))) __bf16 v16bf;
typedef __attribute__((ext_vector_type(8)))  float  v8f;
typedef int v4i __attribute__((vector_size(16)));   // matches builtin param type

// gfx1250 async global->LDS copy (ASYNCcnt-tracked), guarded so absence of the
// builtin falls back to a synchronous vector copy without breaking the build.
#if __has_builtin(__builtin_amdgcn_global_load_async_to_lds_b128) && \
    __has_builtin(__builtin_amdgcn_s_wait_asynccnt)
#define USE_ASYNC_LDS 1
#else
#define USE_ASYNC_LDS 0
#endif

__device__ inline float sigf(float x) { return 1.0f / (1.0f + __expf(-x)); }

// Build a 16x32 bf16 A-fragment from a row-major source.
// Lane layout (ISA 7.12.2, 16-bit A 16x32):
//   lanes 0-15 (M=lane)    : VGPR0-3 = K 0..7,  VGPR4-7 = K 16..23
//   lanes 16-31 (M=lane-16): VGPR0-3 = K 8..15, VGPR4-7 = K 24..31
__device__ inline v16bf load_a_frag(const __bf16* p, int khalf8) {
  v8bf lo = *(const v8bf*)(p + khalf8);
  v8bf hi = *(const v8bf*)(p + 16 + khalf8);
  v16bf a;
#pragma unroll
  for (int i = 0; i < 8; ++i) { a[i] = lo[i]; a[i + 8] = hi[i]; }
  return a;
}

// Pre-packed B tile: 32 lanes x 16 contiguous bf16 (32B per lane, 1KB per tile)
__device__ inline v16bf load_b_tile(const __bf16* packed, int tile, int lane) {
  return *(const v16bf*)(packed + (size_t)tile * 512 + lane * 16);
}

__device__ inline v8f wmma_bf16(v16bf a, v16bf b, v8f c) {
  return __builtin_amdgcn_wmma_f32_16x16x32_bf16(false, a, false, b, (short)0, c,
                                                 false, false);
}

// ---------------------------------------------------------------------------
// Prep: pack fp32 weight W[Ncols x K] into WMMA-B layout (B[k][n] = W[n][k]).
// ---------------------------------------------------------------------------
__global__ void pack_wmma_b(const float* __restrict__ W, __bf16* __restrict__ dst,
                            int Ncols, int K) {
  int KT = K >> 5;
  int total = (Ncols >> 4) * KT * 32;
  int tid = blockIdx.x * blockDim.x + threadIdx.x;
  if (tid >= total) return;
  int lane = tid & 31;
  int tile = tid >> 5;
  int kt = tile % KT;
  int nt = tile / KT;
  int n = nt * 16 + (lane & 15);
  int kbase = kt * 32 + ((lane >> 4) << 3);
  __bf16* out = dst + (size_t)tile * 512 + lane * 16;
  const float* src = W + (size_t)n * K + kbase;
#pragma unroll
  for (int e = 0; e < 16; ++e) out[e] = (__bf16)src[e < 8 ? e : e + 8];
}

__global__ void add_vec(const float* a, const float* b, float* c, int n) {
  int i = blockIdx.x * blockDim.x + threadIdx.x;
  if (i < n) c[i] = a[i] + b[i];
}

// ---------------------------------------------------------------------------
// Bidirectional masked LSTM: one 16-row tile per block, 8 waves x 32 h-cols.
// ---------------------------------------------------------------------------
__global__ __launch_bounds__(256) void lstm_kernel(
    const int* __restrict__ tokens, const int* __restrict__ seg_lens,
    const float* __restrict__ seg_emb, const __bf16* __restrict__ pWih,
    const __bf16* __restrict__ pWhh, const float* __restrict__ bias,
    __bf16* __restrict__ seg_enc, int reverse, int col_off) {
  __shared__ __bf16 xs[16 * Lq * Eq];      // 32 KB: all 16 steps of x for 16 rows
  __shared__ __bf16 hb[2][16 * Hq];        // 16 KB: ping-pong h
  __shared__ int toks[16 * Lq];
  __shared__ int lens[16];

  const int tid = threadIdx.x;
  const int rowbase = blockIdx.x * 16;

  if (tid < 16) lens[tid] = seg_lens[rowbase + tid];
  toks[tid] = tokens[rowbase * Lq + tid];          // 256 tokens, 1/thread
  __syncthreads();
  // Embed: x[r][l][e] = seg_emb[tok][e]
  for (int i = tid; i < 16 * Lq * Eq; i += 256) {
    int e = i & 63;
    xs[i] = (__bf16)seg_emb[toks[i >> 6] * Eq + e];
  }
  for (int i = tid; i < 2 * 16 * Hq; i += 256) ((__bf16*)hb)[i] = (__bf16)0.0f;
  __builtin_prefetch(pWhh + (size_t)tid * 2048, 0, 1);
  __syncthreads();

  const int wave = tid >> 5, lane = tid & 31;
  const int m = lane & 15;                 // A row / C col
  const int khalf8 = (lane >> 4) << 3;     // A lane K-half
  const int mhi = (lane >> 4) << 3;        // C row offset

  v8f c_acc[2], h_reg[2];
#pragma unroll
  for (int j = 0; j < 2; ++j)
#pragma unroll
    for (int r = 0; r < 8; ++r) { c_acc[j][r] = 0.0f; h_reg[j][r] = 0.0f; }

  int cur = 0;
#pragma unroll 1
  for (int t = 0; t < Lq; ++t) {
    const int tt = reverse ? (Lq - 1 - t) : t;
    const __bf16* xrow = xs + m * (Lq * Eq) + tt * Eq;
    const __bf16* hrow = hb[cur] + m * Hq;

#pragma unroll 1
    for (int j = 0; j < 2; ++j) {
      const int cj = wave * 32 + j * 16;
      const int nt0 = cj >> 4;             // tile col base (adds g*16 per gate)
      v8f acc[4];
#pragma unroll
      for (int g = 0; g < 4; ++g) {
        float bv = bias[g * Hq + cj + m];
#pragma unroll
        for (int r = 0; r < 8; ++r) acc[g][r] = bv;
      }
      // x @ Wih^T  (K = 64 -> 2 chunks, A reused across 4 gates)
#pragma unroll 1
      for (int kt = 0; kt < 2; ++kt) {
        v16bf a = load_a_frag(xrow + kt * 32, khalf8);
#pragma unroll
        for (int g = 0; g < 4; ++g)
          acc[g] = wmma_bf16(a, load_b_tile(pWih, (nt0 + g * 16) * 2 + kt, lane),
                             acc[g]);
      }
      // h @ Whh^T  (K = 256 -> 8 chunks)
#pragma unroll 1
      for (int kc = 0; kc < 8; ++kc) {
        v16bf a = load_a_frag(hrow + kc * 32, khalf8);
#pragma unroll
        for (int g = 0; g < 4; ++g)
          acc[g] = wmma_bf16(a, load_b_tile(pWhh, (nt0 + g * 16) * 8 + kc, lane),
                             acc[g]);
      }
      // gates i,f,g,o; freeze (h,c) on masked steps
#pragma unroll
      for (int r = 0; r < 8; ++r) {
        int M = r + mhi;
        bool valid = tt < lens[M];
        float iv = sigf(acc[0][r]);
        float fv = sigf(acc[1][r]);
        float gv = tanhf(acc[2][r]);
        float ov = sigf(acc[3][r]);
        float cn = fv * c_acc[j][r] + iv * gv;
        float hn = ov * tanhf(cn);
        if (valid) { c_acc[j][r] = cn; h_reg[j][r] = hn; }
        hb[cur ^ 1][M * Hq + cj + m] = (__bf16)h_reg[j][r];
      }
    }
    __syncthreads();
    cur ^= 1;
  }
  // final hidden -> seg_enc (bf16, row stride R=512)
#pragma unroll 1
  for (int j = 0; j < 2; ++j) {
    const int cj = wave * 32 + j * 16;
#pragma unroll
    for (int r = 0; r < 8; ++r) {
      int M = r + mhi;
      seg_enc[(size_t)(rowbase + M) * Rq + col_off + cj + m] = (__bf16)h_reg[j][r];
    }
  }
}

// ---------------------------------------------------------------------------
// dec_in[n] = [seg_enc[n], lab_emb[prev_label(n)]]  (bf16, width 576)
// ---------------------------------------------------------------------------
__global__ void build_decin(const __bf16* __restrict__ seg_enc,
                            const int* __restrict__ labels,
                            const float* __restrict__ lab_emb,
                            __bf16* __restrict__ dec_in) {
  long long idx = (long long)blockIdx.x * 256 + threadIdx.x;
  const long long total = (long long)Nq * DIN;
  if (idx >= total) return;
  int k = (int)(idx % DIN);
  long long n = idx / DIN;
  __bf16 v;
  if (k < Rq) {
    v = seg_enc[n * Rq + k];
  } else {
    int b = (int)(n >> 6), t = (int)(n & 63);
    int prev = (t == 0) ? NLq : labels[b * Tq + t - 1];
    v = (__bf16)lab_emb[prev * LEq + (k - Rq)];
  }
  dec_in[idx] = v;
}

// ---------------------------------------------------------------------------
// GRU decoder: 16 batch rows per block (rows independent), 64 steps.
// Step-t dec_in rows staged global->LDS (async on gfx1250), then one 16-col
// subtile at a time with 4 accumulators live (r, z, gi_n, gh_n).
// ---------------------------------------------------------------------------
__global__ __launch_bounds__(256) void gru_kernel(
    const __bf16* __restrict__ dec_in, const __bf16* __restrict__ pWih,
    const __bf16* __restrict__ pWhh, const float* __restrict__ gbih,
    const float* __restrict__ gbhh, float* __restrict__ hs) {
  __shared__ __bf16 hb[2][16 * Rq];        // 32 KB ping-pong h
  __shared__ __bf16 xbuf[16 * DIN];        // 18 KB: current step's inputs
  const int tid = threadIdx.x;
  const int rowbase = blockIdx.x * 16;     // batch rows
  for (int i = tid; i < 2 * 16 * Rq; i += 256) ((__bf16*)hb)[i] = (__bf16)0.0f;
  __builtin_prefetch(pWhh + (size_t)tid * 2048, 0, 1);
  __syncthreads();

  const int wave = tid >> 5, lane = tid & 31;
  const int m = lane & 15;
  const int khalf8 = (lane >> 4) << 3;
  const int mhi = (lane >> 4) << 3;

  v8f h_reg[4];
#pragma unroll
  for (int s = 0; s < 4; ++s)
#pragma unroll
    for (int r = 0; r < 8; ++r) h_reg[s][r] = 0.0f;

  int cur = 0;
#pragma unroll 1
  for (int t = 0; t < Tq; ++t) {
    // ---- stage this step's 16 x 576 bf16 input rows into LDS --------------
    // 72 16-byte chunks per row, 1152 total; ASYNCcnt-tracked on gfx1250.
#pragma unroll 1
    for (int c = tid; c < 16 * (DIN / 8); c += 256) {
      int row = c / (DIN / 8);
      int within = c % (DIN / 8);
      const __bf16* gsrc =
          dec_in + (size_t)((rowbase + row) * Tq + t) * DIN + within * 8;
      __bf16* ldst = xbuf + row * DIN + within * 8;
#if USE_ASYNC_LDS
      __builtin_amdgcn_global_load_async_to_lds_b128(
          (v4i*)(void*)gsrc, (v4i*)(void*)ldst, 0, 0);
#else
      *(v8bf*)ldst = *(const v8bf*)gsrc;
#endif
    }
#if USE_ASYNC_LDS
    __builtin_amdgcn_s_wait_asynccnt(0);
#endif
    __syncthreads();

    const __bf16* xrow = xbuf + m * DIN;
    const __bf16* hrow = hb[cur] + m * Rq;

#pragma unroll 1
    for (int s = 0; s < 4; ++s) {
      const int cbase = wave * 64 + s * 16;  // this wave's 16-col subtile
      const int ntc = cbase >> 4;            // 0..31 within a gate block
      v8f ar, az, agin, aghn;
      {
        int c = cbase + m;
        float br = gbih[c] + gbhh[c];
        float bz = gbih[Rq + c] + gbhh[Rq + c];
        float bi = gbih[2 * Rq + c];
        float bh = gbhh[2 * Rq + c];
#pragma unroll
        for (int r = 0; r < 8; ++r) { ar[r] = br; az[r] = bz; agin[r] = bi; aghn[r] = bh; }
      }
      // input projection: K = 576 -> 18 chunks (gates r, z, gi_n)
#pragma unroll 1
      for (int kc = 0; kc < 18; ++kc) {
        v16bf a = load_a_frag(xrow + kc * 32, khalf8);
        ar   = wmma_bf16(a, load_b_tile(pWih, (ntc)      * 18 + kc, lane), ar);
        az   = wmma_bf16(a, load_b_tile(pWih, (32 + ntc) * 18 + kc, lane), az);
        agin = wmma_bf16(a, load_b_tile(pWih, (64 + ntc) * 18 + kc, lane), agin);
      }
      // recurrent projection: K = 512 -> 16 chunks (gates r, z, gh_n)
#pragma unroll 1
      for (int kc = 0; kc < 16; ++kc) {
        v16bf a = load_a_frag(hrow + kc * 32, khalf8);
        ar   = wmma_bf16(a, load_b_tile(pWhh, (ntc)      * 16 + kc, lane), ar);
        az   = wmma_bf16(a, load_b_tile(pWhh, (32 + ntc) * 16 + kc, lane), az);
        aghn = wmma_bf16(a, load_b_tile(pWhh, (64 + ntc) * 16 + kc, lane), aghn);
      }
#pragma unroll
      for (int r = 0; r < 8; ++r) {
        float rr = sigf(ar[r]);
        float zz = sigf(az[r]);
        float nv = tanhf(agin[r] + rr * aghn[r]);
        float hn = (1.0f - zz) * nv + zz * h_reg[s][r];
        h_reg[s][r] = hn;
        int M = r + mhi;
        hb[cur ^ 1][M * Rq + cbase + m] = (__bf16)hn;
        hs[((size_t)t * Bq + rowbase + M) * Rq + cbase + m] = hn;  // hs[t][b][r]
      }
    }
    __syncthreads();
    cur ^= 1;
  }
}

// ---------------------------------------------------------------------------
// logits[b][t][l] = bout[l] + hs[t][b][:] . Wout[l][:]
// ---------------------------------------------------------------------------
__global__ void logits_kernel(const float* __restrict__ hs,
                              const float* __restrict__ Wout,
                              const float* __restrict__ bout,
                              float* __restrict__ out) {
  int idx = blockIdx.x * 256 + threadIdx.x;
  if (idx >= Bq * Tq * NLq) return;
  int l = idx % NLq;
  int n = idx / NLq;           // n = b*T + t
  int t = n % Tq, b = n / Tq;
  const float* h = hs + ((size_t)t * Bq + b) * Rq;
  const float* w = Wout + l * Rq;
  float acc = bout[l];
  for (int k = 0; k < Rq; ++k) acc += h[k] * w[k];
  out[idx] = acc;
}

// ---------------------------------------------------------------------------
extern "C" void kernel_launch(void* const* d_in, const int* in_sizes, int n_in,
                              void* d_out, int out_size, void* d_ws, size_t ws_size,
                              hipStream_t stream) {
  (void)in_sizes; (void)n_in; (void)out_size; (void)ws_size;
  const int*   tokens   = (const int*)d_in[0];
  const int*   seg_lens = (const int*)d_in[1];
  const int*   labels   = (const int*)d_in[2];
  const float* seg_emb  = (const float*)d_in[3];
  const float* Wih_f    = (const float*)d_in[4];
  const float* Whh_f    = (const float*)d_in[5];
  const float* bih_f    = (const float*)d_in[6];
  const float* bhh_f    = (const float*)d_in[7];
  const float* Wih_b    = (const float*)d_in[8];
  const float* Whh_b    = (const float*)d_in[9];
  const float* bih_b    = (const float*)d_in[10];
  const float* bhh_b    = (const float*)d_in[11];
  const float* lab_emb  = (const float*)d_in[12];
  const float* gWih     = (const float*)d_in[13];
  const float* gWhh     = (const float*)d_in[14];
  const float* gbih     = (const float*)d_in[15];
  const float* gbhh     = (const float*)d_in[16];
  const float* Wout     = (const float*)d_in[17];
  const float* bout     = (const float*)d_in[18];

  char* ws = (char*)d_ws;
  size_t off = 0;
  auto carve = [&](size_t bytes) -> void* {
    void* p = ws + off;
    off += (bytes + 255) & ~(size_t)255;
    return p;
  };
  __bf16* pWih_f  = (__bf16*)carve((size_t)64 * 2  * 512 * 2);
  __bf16* pWhh_f  = (__bf16*)carve((size_t)64 * 8  * 512 * 2);
  __bf16* pWih_b  = (__bf16*)carve((size_t)64 * 2  * 512 * 2);
  __bf16* pWhh_b  = (__bf16*)carve((size_t)64 * 8  * 512 * 2);
  __bf16* pgWih   = (__bf16*)carve((size_t)96 * 18 * 512 * 2);
  __bf16* pgWhh   = (__bf16*)carve((size_t)96 * 16 * 512 * 2);
  float*  bias_f  = (float*) carve((size_t)G4H * 4);
  float*  bias_b  = (float*) carve((size_t)G4H * 4);
  __bf16* seg_enc = (__bf16*)carve((size_t)Nq * Rq * 2);
  __bf16* dec_in  = (__bf16*)carve((size_t)Nq * DIN * 2);
  float*  hs      = (float*) carve((size_t)Tq * Bq * Rq * 4);

  // Weight packing + bias merge (no inter-dependencies)
  pack_wmma_b<<<(64 * 2 * 32 + 255) / 256, 256, 0, stream>>>(Wih_f, pWih_f, G4H, Eq);
  pack_wmma_b<<<(64 * 8 * 32 + 255) / 256, 256, 0, stream>>>(Whh_f, pWhh_f, G4H, Hq);
  pack_wmma_b<<<(64 * 2 * 32 + 255) / 256, 256, 0, stream>>>(Wih_b, pWih_b, G4H, Eq);
  pack_wmma_b<<<(64 * 8 * 32 + 255) / 256, 256, 0, stream>>>(Whh_b, pWhh_b, G4H, Hq);
  pack_wmma_b<<<(96 * 18 * 32 + 255) / 256, 256, 0, stream>>>(gWih, pgWih, G3R, DIN);
  pack_wmma_b<<<(96 * 16 * 32 + 255) / 256, 256, 0, stream>>>(gWhh, pgWhh, G3R, Rq);
  add_vec<<<(G4H + 255) / 256, 256, 0, stream>>>(bih_f, bhh_f, bias_f, G4H);
  add_vec<<<(G4H + 255) / 256, 256, 0, stream>>>(bih_b, bhh_b, bias_b, G4H);

  // Bidirectional LSTM (1024 blocks x 256 threads each)
  lstm_kernel<<<Nq / 16, 256, 0, stream>>>(tokens, seg_lens, seg_emb, pWih_f,
                                           pWhh_f, bias_f, seg_enc, 0, 0);
  lstm_kernel<<<Nq / 16, 256, 0, stream>>>(tokens, seg_lens, seg_emb, pWih_b,
                                           pWhh_b, bias_b, seg_enc, 1, Hq);

  // Decoder input assembly
  long long din_total = (long long)Nq * DIN;
  build_decin<<<(unsigned)((din_total + 255) / 256), 256, 0, stream>>>(
      seg_enc, labels, lab_emb, dec_in);

  // GRU decoder (16 blocks, rows independent)
  gru_kernel<<<Bq / 16, 256, 0, stream>>>(dec_in, pgWih, pgWhh, gbih, gbhh, hs);

  // Output projection
  logits_kernel<<<(Bq * Tq * NLq + 255) / 256, 256, 0, stream>>>(
      hs, Wout, bout, (float*)d_out);
}
